// SubAttention_8057358647537
// MI455X (gfx1250) — matrix operand order repaired and checked by reading, not compiled, not run on hardware
//
#include <hip/hip_runtime.h>
#include <hip/hip_bf16.h>

// ---------------------------------------------------------------------------
// CDNA5 (gfx1250, wave32) implementation of the linear-attention block.
// GEMM stages: v_wmma_f32_16x16x32_bf16 (bf16 operands, fp32 accum).
// A-tile staging in the projection GEMMs uses the Tensor Data Mover
// (tensor_load_to_lds + s_wait_tensorcnt) with double buffering when the
// builtin is available; otherwise a cooperative LDS load path.
// ---------------------------------------------------------------------------

typedef __bf16 bf16_t;
typedef __attribute__((ext_vector_type(16))) __bf16 v16bf;
typedef __attribute__((ext_vector_type(8)))  __bf16 v8bf;
typedef __attribute__((ext_vector_type(4)))  __bf16 v4bf;
typedef __attribute__((ext_vector_type(8)))  float  v8f;
typedef __attribute__((ext_vector_type(4)))  float  v4f;

#define NHEADS 8
#define DHEAD  128
#define NEGC   (-1000000.0f)
#define LN_EPS 1e-5f

#if defined(__HIP_DEVICE_COMPILE__) && defined(__gfx1250__) && \
    __has_builtin(__builtin_amdgcn_tensor_load_to_lds) &&      \
    __has_builtin(__builtin_amdgcn_s_wait_tensorcnt)
#define HAVE_TDM 1
#else
#define HAVE_TDM 0
#endif

#if HAVE_TDM
typedef __attribute__((ext_vector_type(4))) unsigned int v4u;
typedef __attribute__((ext_vector_type(4))) int v4i;
typedef __attribute__((ext_vector_type(8))) int v8i;

// Issue one TDM 2D tile load: tile_d1 rows x tile_d0 (2-byte) elements from a
// row-major tensor with row stride `stride_elems`, into LDS at byte offset
// lds_off.  pad: after every 16 DWORDs (one 32-elem row) insert 4 DWORDs,
// producing the 40-element padded LDS rows the WMMA fragment reads expect.
__device__ __forceinline__ void tdm_load_tile_2d(unsigned int lds_off,
                                                 const void* gaddr,
                                                 unsigned int tensor_d0,
                                                 unsigned int tensor_d1,
                                                 unsigned int tile_d0,
                                                 unsigned int tile_d1,
                                                 unsigned int stride_elems) {
  const unsigned long long ga = (unsigned long long)(size_t)gaddr;
  v4u g0;
  g0[0] = 1u;                                       // count=1, user descriptor
  g0[1] = lds_off;                                  // lds_addr (bytes)
  g0[2] = (unsigned int)(ga & 0xFFFFFFFFu);         // global_addr[31:0]
  g0[3] = (unsigned int)((ga >> 32) & 0x1FFFFFFu)   // global_addr[56:32]
          | (2u << 30);                             // type = 2 (image)
  v8i g1;
  g1[0] = (int)((1u << 16)        // data_size = 2 bytes
              | (1u << 20)        // pad_enable
              | (3u << 22)        // pad_interval: 16 DWORDs
              | (3u << 25));      // pad_amount:    4 DWORDs
  g1[1] = (int)((tensor_d0 & 0xFFFFu) << 16);                       // dim0 lo
  g1[2] = (int)(((tensor_d0 >> 16) & 0xFFFFu) |
                ((tensor_d1 & 0xFFFFu) << 16));                     // dim0 hi | dim1 lo
  g1[3] = (int)(((tensor_d1 >> 16) & 0xFFFFu) |
                ((tile_d0 & 0xFFFFu) << 16));                       // dim1 hi | tile0
  g1[4] = (int)(tile_d1 & 0xFFFFu);                                 // tile1 (tile2=0)
  g1[5] = (int)stride_elems;                                        // dim0_stride lo32
  g1[6] = 0;
  g1[7] = 0;
  const v4i gz = {0, 0, 0, 0};
#if __clang_major__ >= 23
  const v8i gz8 = {0, 0, 0, 0, 0, 0, 0, 0};
  __builtin_amdgcn_tensor_load_to_lds(g0, g1, gz, gz, gz8, 0);
#else
  __builtin_amdgcn_tensor_load_to_lds(g0, g1, gz, gz, 0);
#endif
}
#endif  // HAVE_TDM

// Build a 16-element bf16 fragment from two 16-byte runs (16B aligned, avoids
// over-strict 32B vector alignment assumptions on padded LDS rows).
__device__ __forceinline__ v16bf load_frag2(const bf16_t* p0, const bf16_t* p1) {
  v16bf r;
  *((v8bf*)&r)     = *(const v8bf*)p0;
  *((v8bf*)&r + 1) = *(const v8bf*)p1;
  return r;
}

__device__ __forceinline__ v8f wmma_bf16(v16bf a, v16bf b, v8f c) {
  return __builtin_amdgcn_wmma_f32_16x16x32_bf16(false, a, false, b,
                                                 (short)0, c, false, false);
}

// ---------------------------------------------------------------------------
// LayerNorm over last dim, output bf16. One block (256 threads) per row.
// ---------------------------------------------------------------------------
__global__ __launch_bounds__(256)
void ln_bf16_kernel(const float* __restrict__ x, const float* __restrict__ g,
                    const float* __restrict__ b, bf16_t* __restrict__ out, int D) {
  __shared__ float red0[256];
  __shared__ float red1[256];
  const int row = blockIdx.x;
  const float* xr = x + (size_t)row * D;
  bf16_t* orow = out + (size_t)row * D;
  float s = 0.f, ss = 0.f;
  for (int i = threadIdx.x; i < D; i += 256) {
    float v = xr[i];
    s += v; ss += v * v;
  }
  red0[threadIdx.x] = s; red1[threadIdx.x] = ss;
  __syncthreads();
  for (int off = 128; off > 0; off >>= 1) {
    if (threadIdx.x < off) {
      red0[threadIdx.x] += red0[threadIdx.x + off];
      red1[threadIdx.x] += red1[threadIdx.x + off];
    }
    __syncthreads();
  }
  const float mu  = red0[0] / (float)D;
  const float var = red1[0] / (float)D - mu * mu;
  const float inv = rsqrtf(var + LN_EPS);
  for (int i = threadIdx.x; i < D; i += 256) {
    float v = (xr[i] - mu) * inv * g[i] + b[i];
    orow[i] = (bf16_t)v;
  }
}

// ---------------------------------------------------------------------------
// fp32 -> bf16 elementwise (weights).
// ---------------------------------------------------------------------------
__global__ __launch_bounds__(256)
void cvt_bf16_kernel(const float* __restrict__ in, bf16_t* __restrict__ out, int n) {
  int i = blockIdx.x * 256 + threadIdx.x;
  if (i < n) out[i] = (bf16_t)in[i];
}

// ---------------------------------------------------------------------------
// Generic WMMA GEMM: C[M x Dout] = A_bf16[M x K] @ W_bf16[Dout x K]^T + bias.
//   mode 0 (KEY):   out_f32[(b,h,nseq,dd)] = val + (1-mask[m])*NEG
//   mode 1 (VALUE): out_bf16[(b,h,nseq,dd)] = bf16(val * mask[m])
//   mode 2 (PLAIN): out_f32[m*Dout + n] = val
// Block = 256 threads (8 waves), 128x128 output tile. A tile staged into
// padded LDS by the TDM (double-buffered, overlapped with compute) when
// available; W rows stream from L2 as contiguous B-fragments.
// ---------------------------------------------------------------------------
struct GemmParams {
  const bf16_t* A;
  const bf16_t* W;
  const float*  bias;
  const float*  mask;     // (B*Trows,) flat; null for PLAIN
  float*        outF;     // KEY / PLAIN
  bf16_t*       outH;     // VALUE
  int M, K, Dout;
  int Trows;              // rows per batch in A (T or Tc)
  int Nseq;               // concatenated sequence length
  int rowBase;            // 0 for cond part, Tc for x part
  int mode;
};

#define AS_BUF_BYTES (128 * 40 * 2)   // one padded 128x32 bf16 tile = 10240 B

__global__ __launch_bounds__(256)
void wmma_gemm_kernel(GemmParams p) {
  __shared__ __align__(16) bf16_t As[2][128][40];
  const int wave = threadIdx.x >> 5;
  const int lane = threadIdx.x & 31;
  const bool hi  = (lane >= 16);
  const int mBase = blockIdx.y * 128;
  const int nBase = blockIdx.x * 128;

  v8f acc[8] = {};

  const int mrow = wave * 16 + (lane & 15);
  const int aoff = hi ? 8 : 0;                // half of A-frag chunk base within k-tile
  const int boff = hi ? 16 : 0;               // B-frag chunk base within k-tile
  const int nk   = p.K >> 5;                  // number of 32-wide k tiles

#if HAVE_TDM
  // Prologue: DMA tile 0 into buffer 0.
  if (wave == 0) {
    tdm_load_tile_2d(0u, p.A + (size_t)mBase * p.K, (unsigned)p.K,
                     (unsigned)p.M, 32u, 128u, (unsigned)p.K);
  }
#else
  const int ldRow = threadIdx.x >> 1;          // 0..127
  const int ldCol = (threadIdx.x & 1) * 16;    // 0 or 16
#endif

  for (int i = 0; i < nk; ++i) {
    const int buf = i & 1;
    const int k0  = i << 5;
#if HAVE_TDM
    if (wave == 0) __builtin_amdgcn_s_wait_tensorcnt(0);
    __syncthreads();                            // tile i resident in As[buf]
    if (wave == 0 && (i + 1) < nk) {            // overlap DMA(i+1) with compute
      tdm_load_tile_2d((unsigned)((buf ^ 1) * AS_BUF_BYTES),
                       p.A + (size_t)mBase * p.K + (k0 + 32), (unsigned)p.K,
                       (unsigned)p.M, 32u, 128u, (unsigned)p.K);
    }
#else
    {
      const bf16_t* src = p.A + (size_t)(mBase + ldRow) * p.K + (k0 + ldCol);
      *(v8bf*)&As[buf][ldRow][ldCol]     = *(const v8bf*)src;
      *(v8bf*)&As[buf][ldRow][ldCol + 8] = *(const v8bf*)(src + 8);
    }
    __syncthreads();
#endif

    const bf16_t* arow = &As[buf][mrow][0];
    v16bf afrag = load_frag2(arow + 2 * aoff, arow + 2 * aoff + 8);

    // Batch all 8 B fragments (one load clause), then 8 back-to-back WMMAs.
    const int kb = k0 + boff;
    v16bf bfr[8];
#pragma unroll
    for (int nt = 0; nt < 8; ++nt) {
      const int n = nBase + nt * 16 + (lane & 15);
      const bf16_t* wrow = p.W + (size_t)n * p.K + kb;
      bfr[nt] = load_frag2(wrow, wrow + 8);
      __builtin_prefetch((const void*)(wrow + 32), 0, 3);  // next k-tile of W
    }
#pragma unroll
    for (int nt = 0; nt < 8; ++nt) acc[nt] = wmma_bf16(afrag, bfr[nt], acc[nt]);

    __syncthreads();                            // protect As[buf] before reuse
  }

  // epilogue
#pragma unroll
  for (int nt = 0; nt < 8; ++nt) {
    const int n = nBase + nt * 16 + (lane & 15);
    const float bia = p.bias[n];
#pragma unroll
    for (int r = 0; r < 8; ++r) {
      const int m = mBase + wave * 16 + r + (hi ? 8 : 0);
      float val = acc[nt][r] + bia;
      if (p.mode == 2) {
        p.outF[(size_t)m * p.Dout + n] = val;
      } else {
        const int bidx = m / p.Trows;
        const int t    = m - bidx * p.Trows;
        const int nseq = p.rowBase + t;
        const int h    = n >> 7;
        const int dd   = n & 127;
        const size_t oidx =
            (((size_t)bidx * NHEADS + h) * p.Nseq + nseq) * DHEAD + dd;
        const float mk = p.mask[m];
        if (p.mode == 0) p.outF[oidx] = val + (1.0f - mk) * NEGC;
        else             p.outH[oidx] = (bf16_t)(val * mk);
      }
    }
  }
}

// ---------------------------------------------------------------------------
// q softmax over head dim (128). One wave per (b,t,h) row; output bf16 in
// head-major layout (B,H,T,128).
// ---------------------------------------------------------------------------
__global__ __launch_bounds__(256)
void softmax_q_kernel(const float* __restrict__ q, bf16_t* __restrict__ out, int T) {
  const int wave = threadIdx.x >> 5;
  const int lane = threadIdx.x & 31;
  const long rowId = (long)blockIdx.x * 8 + wave;     // over B*T*H
  const int h = (int)(rowId % NHEADS);
  const long bt = rowId / NHEADS;
  const int t = (int)(bt % T);
  const int b = (int)(bt / T);

  const float* src = q + ((size_t)b * T + t) * (NHEADS * DHEAD) + h * DHEAD + lane * 4;
  v4f v = *(const v4f*)src;
  float m = fmaxf(fmaxf(v[0], v[1]), fmaxf(v[2], v[3]));
#pragma unroll
  for (int off = 16; off > 0; off >>= 1) m = fmaxf(m, __shfl_xor(m, off, 32));
  float e0 = __expf(v[0] - m), e1 = __expf(v[1] - m);
  float e2 = __expf(v[2] - m), e3 = __expf(v[3] - m);
  float s = e0 + e1 + e2 + e3;
#pragma unroll
  for (int off = 16; off > 0; off >>= 1) s += __shfl_xor(s, off, 32);
  const float inv = 1.0f / s;
  bf16_t* dst = out + (((size_t)b * NHEADS + h) * T + t) * DHEAD + lane * 4;
  v4bf o = { (bf16_t)(e0 * inv), (bf16_t)(e1 * inv),
             (bf16_t)(e2 * inv), (bf16_t)(e3 * inv) };
  *(v4bf*)dst = o;
}

// ---------------------------------------------------------------------------
// k softmax over the sequence dim N (column softmax). Block = (b,h), 128
// threads = dd; online max+sum, then normalized bf16 write.
// ---------------------------------------------------------------------------
__global__ __launch_bounds__(128)
void softmax_k_kernel(const float* __restrict__ keyF, bf16_t* __restrict__ out,
                      int Nseq) {
  const int bh = blockIdx.x;
  const int dd = threadIdx.x;
  const float* src = keyF + (size_t)bh * Nseq * DHEAD + dd;
  float m = -3.0e38f, s = 0.f;
  for (int n = 0; n < Nseq; ++n) {
    const float v = src[(size_t)n * DHEAD];
    if (v > m) { s *= __expf(m - v); m = v; }
    s += __expf(v - m);
  }
  const float inv = 1.0f / s;
  bf16_t* dst = out + (size_t)bh * Nseq * DHEAD + dd;
  for (int n = 0; n < Nseq; ++n) {
    const float v = src[(size_t)n * DHEAD];
    dst[(size_t)n * DHEAD] = (bf16_t)(__expf(v - m) * inv);
  }
}

// ---------------------------------------------------------------------------
// KV einsum: attn[b,h,d,l] = sum_n k[b,h,n,d] * v[b,h,n,l].
// One block per (b,h); 32xN chunks transposed into padded LDS so both WMMA
// fragments are contiguous. Output stored TRANSPOSED as attnT[l][d] (bf16) so
// the next GEMM's B-fragments are contiguous.
// ---------------------------------------------------------------------------
__global__ __launch_bounds__(256)
void attn_kv_kernel(const bf16_t* __restrict__ kbf, const bf16_t* __restrict__ vbf,
                    bf16_t* __restrict__ attnT, int Nseq) {
  __shared__ __align__(16) bf16_t kT[128][40];
  __shared__ __align__(16) bf16_t vT[128][40];
  const int bh   = blockIdx.x;
  const int wave = threadIdx.x >> 5;
  const int lane = threadIdx.x & 31;
  const bool hi  = (lane >= 16);
  const bf16_t* kbase = kbf + (size_t)bh * Nseq * DHEAD;
  const bf16_t* vbase = vbf + (size_t)bh * Nseq * DHEAD;

  v8f acc[8] = {};
  const int ldr = threadIdx.x >> 3;          // n within chunk (0..31)
  const int ldc = (threadIdx.x & 7) * 16;    // d0
  const int aoff = hi ? 8 : 0;
  const int boff = hi ? 16 : 0;
  const int arow = wave * 16 + (lane & 15);

  for (int nc = 0; nc < Nseq; nc += 32) {
    {
      const bf16_t* ks = kbase + (size_t)(nc + ldr) * DHEAD + ldc;
      const bf16_t* vs = vbase + (size_t)(nc + ldr) * DHEAD + ldc;
      v16bf kv = load_frag2(ks, ks + 8);
      v16bf vv = load_frag2(vs, vs + 8);
#pragma unroll
      for (int j = 0; j < 16; ++j) {
        kT[ldc + j][ldr] = kv[j];
        vT[ldc + j][ldr] = vv[j];
      }
    }
    __syncthreads();

    const bf16_t* ar = &kT[arow][0];
    v16bf afrag = load_frag2(ar + 2 * aoff, ar + 2 * aoff + 8);
    v16bf bfr[8];
#pragma unroll
    for (int nt = 0; nt < 8; ++nt) {
      const bf16_t* br = &vT[nt * 16 + (lane & 15)][boff];
      bfr[nt] = load_frag2(br, br + 8);
    }
#pragma unroll
    for (int nt = 0; nt < 8; ++nt) acc[nt] = wmma_bf16(afrag, bfr[nt], acc[nt]);
    __syncthreads();
  }

  bf16_t* abase = attnT + (size_t)bh * DHEAD * DHEAD;
#pragma unroll
  for (int nt = 0; nt < 8; ++nt) {
#pragma unroll
    for (int r = 0; r < 8; ++r) {
      const int d = wave * 16 + r + (hi ? 8 : 0);
      const int l = nt * 16 + (lane & 15);
      abase[(size_t)l * DHEAD + d] = (bf16_t)acc[nt][r];   // transposed store
    }
  }
}

// ---------------------------------------------------------------------------
// y[b,h,t,l] = sum_d q[b,h,t,d] * attn[b,h,d,l], using attnT (l,d) rows as
// contiguous B-fragments. Output interleaved into y_bf16 (B,T,D).
// ---------------------------------------------------------------------------
__global__ __launch_bounds__(256)
void y_gemm_kernel(const bf16_t* __restrict__ qbf, const bf16_t* __restrict__ attnT,
                   bf16_t* __restrict__ ybf, int T) {
  const int bh     = blockIdx.y;                  // B*H
  const int rowBlk = blockIdx.x;                  // T/128
  const int wave = threadIdx.x >> 5;
  const int lane = threadIdx.x & 31;
  const bool hi  = (lane >= 16);
  const bf16_t* qb = qbf   + (size_t)bh * T * DHEAD;
  const bf16_t* ab = attnT + (size_t)bh * DHEAD * DHEAD;
  const int m = rowBlk * 128 + wave * 16 + (lane & 15);
  v8f acc[8] = {};

#pragma unroll
  for (int kk = 0; kk < DHEAD; kk += 32) {
    const int ao = kk + (hi ? 16 : 0);
    v16bf afrag = load_frag2(qb + (size_t)m * DHEAD + ao,
                             qb + (size_t)m * DHEAD + ao + 8);
    const int bo = kk + (hi ? 16 : 0);
    v16bf bfr[8];
#pragma unroll
    for (int nt = 0; nt < 8; ++nt) {
      const bf16_t* br = ab + (size_t)(nt * 16 + (lane & 15)) * DHEAD + bo;
      bfr[nt] = load_frag2(br, br + 8);
    }
#pragma unroll
    for (int nt = 0; nt < 8; ++nt) acc[nt] = wmma_bf16(afrag, bfr[nt], acc[nt]);
  }

  const int b = bh >> 3, h = bh & 7;
#pragma unroll
  for (int nt = 0; nt < 8; ++nt) {
#pragma unroll
    for (int r = 0; r < 8; ++r) {
      const int t = rowBlk * 128 + wave * 16 + r + (hi ? 8 : 0);
      const int l = nt * 16 + (lane & 15);
      ybf[((size_t)b * T + t) * (NHEADS * DHEAD) + h * DHEAD + l] = (bf16_t)acc[nt][r];
    }
  }
}

// ---------------------------------------------------------------------------
// Host-side orchestration.
// ---------------------------------------------------------------------------
extern "C" void kernel_launch(void* const* d_in, const int* in_sizes, int n_in,
                              void* d_out, int out_size, void* d_ws, size_t ws_size,
                              hipStream_t stream) {
  const int B = 8, T = 2048, Tc = 512, D = 1024, Dc = 512;
  const int N = T + Tc;

  const float* query     = (const float*)d_in[0];
  const float* x         = (const float*)d_in[1];
  const float* cond_emb  = (const float*)d_in[2];
  const float* src_mask  = (const float*)d_in[3];
  const float* cond_mask = (const float*)d_in[4];
  const float* norm_x_g  = (const float*)d_in[5];
  const float* norm_x_b  = (const float*)d_in[6];
  const float* norm_c_g  = (const float*)d_in[7];
  const float* norm_c_b  = (const float*)d_in[8];
  const float* W_kc = (const float*)d_in[9];
  const float* b_kc = (const float*)d_in[10];
  const float* W_vc = (const float*)d_in[11];
  const float* b_vc = (const float*)d_in[12];
  const float* W_kx = (const float*)d_in[13];
  const float* b_kx = (const float*)d_in[14];
  const float* W_vx = (const float*)d_in[15];
  const float* b_vx = (const float*)d_in[16];
  const float* W_y  = (const float*)d_in[17];
  const float* b_y  = (const float*)d_in[18];

  // ---- workspace carve-out (~250 MB assumed available) ----
  char* ws = (char*)d_ws;
  size_t off = 0;
  auto carve = [&](size_t bytes) -> void* {
    void* p = ws + off;
    off = (off + bytes + 255) & ~(size_t)255;
    return p;
  };
  bf16_t* xn_bf  = (bf16_t*)carve((size_t)B * T * D * 2);        // 32 MB
  bf16_t* cn_bf  = (bf16_t*)carve((size_t)B * Tc * Dc * 2);      //  4 MB
  bf16_t* Wkc_bf = (bf16_t*)carve((size_t)D * Dc * 2);
  bf16_t* Wvc_bf = (bf16_t*)carve((size_t)D * Dc * 2);
  bf16_t* Wkx_bf = (bf16_t*)carve((size_t)D * D * 2);
  bf16_t* Wvx_bf = (bf16_t*)carve((size_t)D * D * 2);
  bf16_t* Wy_bf  = (bf16_t*)carve((size_t)D * D * 2);
  float*  keyF   = (float*) carve((size_t)B * NHEADS * N * DHEAD * 4);  // 80 MB
  bf16_t* k_bf   = (bf16_t*)carve((size_t)B * NHEADS * N * DHEAD * 2);  // 40 MB
  bf16_t* v_bf   = (bf16_t*)carve((size_t)B * NHEADS * N * DHEAD * 2);  // 40 MB
  bf16_t* q_bf   = (bf16_t*)carve((size_t)B * NHEADS * T * DHEAD * 2);  // 32 MB
  bf16_t* attnT  = (bf16_t*)carve((size_t)B * NHEADS * DHEAD * DHEAD * 2);
  bf16_t* y_bf   = (bf16_t*)keyF;   // alias: keyF dead after softmax_k

  // 1) LayerNorms -> bf16
  ln_bf16_kernel<<<B * T, 256, 0, stream>>>(x, norm_x_g, norm_x_b, xn_bf, D);
  ln_bf16_kernel<<<B * Tc, 256, 0, stream>>>(cond_emb, norm_c_g, norm_c_b, cn_bf, Dc);

  // 2) Weight conversions
  auto cvt = [&](const float* src, bf16_t* dst, int n) {
    cvt_bf16_kernel<<<(n + 255) / 256, 256, 0, stream>>>(src, dst, n);
  };
  cvt(W_kc, Wkc_bf, D * Dc);
  cvt(W_vc, Wvc_bf, D * Dc);
  cvt(W_kx, Wkx_bf, D * D);
  cvt(W_vx, Wvx_bf, D * D);
  cvt(W_y,  Wy_bf,  D * D);

  // 3) Key / value projections (write head-major (B,H,N,128))
  auto run_gemm = [&](const bf16_t* A, const bf16_t* W, const float* bias,
                      const float* mask, float* outF, bf16_t* outH,
                      int M, int K, int Dout, int Trows, int rowBase, int mode) {
    GemmParams p{A, W, bias, mask, outF, outH, M, K, Dout, Trows, N, rowBase, mode};
    dim3 grid(Dout / 128, M / 128);
    wmma_gemm_kernel<<<grid, 256, 0, stream>>>(p);
  };
  run_gemm(cn_bf, Wkc_bf, b_kc, cond_mask, keyF, nullptr,
           B * Tc, Dc, D, Tc, /*rowBase=*/0,  /*mode=*/0);   // k_cond
  run_gemm(xn_bf, Wkx_bf, b_kx, src_mask,  keyF, nullptr,
           B * T,  D,  D, T,  /*rowBase=*/Tc, /*mode=*/0);   // k_x
  run_gemm(cn_bf, Wvc_bf, b_vc, cond_mask, nullptr, v_bf,
           B * Tc, Dc, D, Tc, /*rowBase=*/0,  /*mode=*/1);   // v_cond
  run_gemm(xn_bf, Wvx_bf, b_vx, src_mask,  nullptr, v_bf,
           B * T,  D,  D, T,  /*rowBase=*/Tc, /*mode=*/1);   // v_x

  // 4) Softmaxes
  softmax_q_kernel<<<(B * T * NHEADS) / 8, 256, 0, stream>>>(query, q_bf, T);
  softmax_k_kernel<<<B * NHEADS, 128, 0, stream>>>(keyF, k_bf, N);

  // 5) KV einsum -> attnT (bf16, transposed)
  attn_kv_kernel<<<B * NHEADS, 256, 0, stream>>>(k_bf, v_bf, attnT, N);

  // 6) y = q @ attn (per head), interleave into (B,T,D) bf16
  {
    dim3 grid(T / 128, B * NHEADS);
    y_gemm_kernel<<<grid, 256, 0, stream>>>(q_bf, attnT, y_bf, T);
  }

  // 7) Final projection: out = y @ W_y^T + b_y (fp32 out)
  run_gemm(y_bf, Wy_bf, b_y, nullptr, (float*)d_out, nullptr,
           B * T, D, D, T, 0, /*mode=*/2);
}
